// LFA_23484881174679
// MI455X (gfx1250) — matrix operand order repaired and checked by reading, not compiled, not run on hardware
//
#include <hip/hip_runtime.h>
#include <hip/hip_bf16.h>
#include <math.h>

typedef __bf16 bf16;
typedef __attribute__((ext_vector_type(16))) __bf16 v16bf;
typedef __attribute__((ext_vector_type(8)))  __bf16 v8bf;
typedef __attribute__((ext_vector_type(8)))  float  v8f;

#define EPS_   1e-5f
#define SLOPE_ 0.1f

static constexpr int Bb   = 2;
static constexpr int Nn   = 40960;
static constexpr int Kk   = 16;
static constexpr int Cc   = 64;
static constexpr int OUTc = 64;
static constexpr int CIN  = 74;          // C + 10
static constexpr int CPAD = 96;          // padded to 3 x 32 for the K-dim of WMMA
static constexpr int NPTS = Bb * Nn;     // 81920

__device__ __forceinline__ float lrelu(float v) { return v >= 0.f ? v : SLOPE_ * v; }

__device__ __forceinline__ v8f wmma_bf16(v16bf a, v16bf b, v8f c) {
  // (neg_a, A, neg_b, B, c_mod, C, reuse_a, reuse_b)
  return __builtin_amdgcn_wmma_f32_16x16x32_bf16(false, a, false, b, (short)0, c, false, false);
}

// 16-bit A 16x32 fragment, contiguous k order (not ISA table order): lane L -> row m=L%16, half=L/16,
// element e -> K = e + 16*half
__device__ __forceinline__ v16bf load_a_frag(const bf16* w, int row0, int ldw, int kc, int lane) {
  const int m = lane & 15, half = lane >> 4;
  const bf16* base = w + (size_t)(row0 + m) * ldw + kc * 32 + 16 * half;
  v16bf f;
#pragma unroll
  for (int e = 0; e < 16; ++e) f[e] = base[e];
  return f;
}

__device__ __forceinline__ void wave_lds_fence() {
  asm volatile("" ::: "memory");
  __builtin_amdgcn_wave_barrier();
  asm volatile("" ::: "memory");
}

// --------------------------------------------------------------------------
// Zero the stats accumulators (first 512 floats of workspace)
__global__ __launch_bounds__(512) void k_zero(float* ws) {
  if (threadIdx.x < 512) ws[threadIdx.x] = 0.f;
}

// --------------------------------------------------------------------------
// Kernel A: per point, build 96x16 bf16 panel (feat gather + rel-pos),
// x1 = W1 @ panel via 12 WMMAs, store x1 col-major (bf16), accumulate stats.
__global__ __launch_bounds__(256) void k_gemm1(
    const float* __restrict__ xyz, const float* __restrict__ feat,
    const int* __restrict__ nidx, const float* __restrict__ W1,
    bf16* __restrict__ x1, float* __restrict__ stats1)
{
  __shared__ alignas(64) bf16 w1s[OUTc * CPAD];     // 12 KB
  __shared__ alignas(64) bf16 panel[8][Kk * CPAD];  // 24 KB, col-major [k][row]

  const int lane   = threadIdx.x & 31;
  const int wave   = threadIdx.x >> 5;
  const int gwave  = (int)((blockIdx.x * blockDim.x + threadIdx.x) >> 5);
  const int nwaves = (int)((gridDim.x * blockDim.x) >> 5);
  const int half = lane >> 4, c = lane & 15;

  // W1 (64x74) -> LDS, zero-padded to 64x96
  for (int i = threadIdx.x; i < OUTc * CPAD; i += blockDim.x) {
    int r = i / CPAD, cc = i % CPAD;
    w1s[i] = (cc < CIN) ? (bf16)W1[r * CIN + cc] : (bf16)0.0f;
  }
  // zero my wave's panel once (rows 74..95 stay zero forever)
  for (int i = lane; i < Kk * CPAD; i += 32) panel[wave][i] = (bf16)0.0f;
  __syncthreads();

  // hoist A fragments (4 row-tiles x 3 k-chunks)
  v16bf A[4][3];
#pragma unroll
  for (int t = 0; t < 4; ++t)
#pragma unroll
    for (int kc = 0; kc < 3; ++kc) A[t][kc] = load_a_frag(w1s, t * 16, CPAD, kc, lane);

  float ssum[4][8] = {}, ssq[4][8] = {};

  for (int pt = gwave; pt < NPTS; pt += nwaves) {
    const int b = pt / Nn;
    // each lane owns neighbor k = lane&15 for the gather
    int iv = (lane < Kk) ? nidx[(size_t)pt * Kk + lane] : 0;
    int myidx = __shfl(iv, lane & 15, 32);

    // relative-position rows 64..73 (computed by lanes 0..15, one per neighbor)
    if (lane < Kk) {
      const float cx = xyz[(size_t)pt * 3 + 0];
      const float cy = xyz[(size_t)pt * 3 + 1];
      const float cz = xyz[(size_t)pt * 3 + 2];
      const float nx = xyz[((size_t)b * Nn + iv) * 3 + 0];
      const float ny = xyz[((size_t)b * Nn + iv) * 3 + 1];
      const float nz = xyz[((size_t)b * Nn + iv) * 3 + 2];
      const float dx = cx - nx, dy = cy - ny, dz = cz - nz;
      const float dist = sqrtf(dx * dx + dy * dy + dz * dz);
      bf16* col = &panel[wave][lane * CPAD + Cc];
      col[0] = (bf16)dx;  col[1] = (bf16)dy;  col[2] = (bf16)dz; col[3] = (bf16)dist;
      col[4] = (bf16)cx;  col[5] = (bf16)cy;  col[6] = (bf16)cz;
      col[7] = (bf16)nx;  col[8] = (bf16)ny;  col[9] = (bf16)nz;
    }
    // feature gather rows 0..63: lane's k fixed (=c), channels 32*half + i,
    // packed 4-at-a-time into ds_store_b64
    const float* fb = feat + (size_t)b * Cc * Nn;
#pragma unroll
    for (int i = 0; i < 32; i += 4) {
      const int ch = 32 * half + i;
      union { bf16 h[4]; uint2 u; } pk;
      pk.h[0] = (bf16)fb[(size_t)(ch + 0) * Nn + myidx];
      pk.h[1] = (bf16)fb[(size_t)(ch + 1) * Nn + myidx];
      pk.h[2] = (bf16)fb[(size_t)(ch + 2) * Nn + myidx];
      pk.h[3] = (bf16)fb[(size_t)(ch + 3) * Nn + myidx];
      *(uint2*)(&panel[wave][c * CPAD + ch]) = pk.u;
    }
    wave_lds_fence();

    // B fragments: contiguous k order matching A (not ISA table order): lane -> col=c, element e -> K=e+16*half
    v16bf Bf[3];
#pragma unroll
    for (int kc = 0; kc < 3; ++kc)
      Bf[kc] = *(const v16bf*)(&panel[wave][c * CPAD + kc * 32 + 16 * half]);

    v8f acc[4];
#pragma unroll
    for (int t = 0; t < 4; ++t) {
      v8f a = {};
      a = wmma_bf16(A[t][0], Bf[0], a);
      a = wmma_bf16(A[t][1], Bf[1], a);
      a = wmma_bf16(A[t][2], Bf[2], a);
      acc[t] = a;
    }

    // store x1 col-major [pt][col*64+row] as packed b128 + accumulate stats
    bf16* xo = x1 + (size_t)pt * (OUTc * Kk);
#pragma unroll
    for (int t = 0; t < 4; ++t) {
      union { bf16 h[8]; uint4 u; } pk;
#pragma unroll
      for (int r = 0; r < 8; ++r) {
        float v = acc[t][r];
        ssum[t][r] += v;
        ssq[t][r] += v * v;
        pk.h[r] = (bf16)v;
      }
      *(uint4*)(xo + c * 64 + t * 16 + 8 * half) = pk.u;
    }
  }

  // flush stats: reduce across the 16 columns then atomicAdd per channel
#pragma unroll
  for (int t = 0; t < 4; ++t)
#pragma unroll
    for (int r = 0; r < 8; ++r) {
      float s = ssum[t][r], q = ssq[t][r];
      for (int m = 8; m >= 1; m >>= 1) { s += __shfl_xor(s, m, 16); q += __shfl_xor(q, m, 16); }
      if (c == 0) {
        int row = t * 16 + half * 8 + r;
        atomicAdd(&stats1[row], s);
        atomicAdd(&stats1[64 + row], q);
      }
    }
}

// --------------------------------------------------------------------------
// Finalize BN: aff[c] = g*invstd, aff[64+c] = b - mean*invstd*g
__global__ __launch_bounds__(64) void k_fin(const float* __restrict__ stats,
                                            const float* __restrict__ g,
                                            const float* __restrict__ bb,
                                            float* __restrict__ aff, float cnt)
{
  int ch = threadIdx.x;
  if (ch >= 64) return;
  float mean = stats[ch] / cnt;
  float var  = stats[64 + ch] / cnt - mean * mean;
  float inv  = rsqrtf(var + EPS_);
  aff[ch]      = g[ch] * inv;
  aff[64 + ch] = bb[ch] - mean * inv * g[ch];
}

// --------------------------------------------------------------------------
// Kernel C: xn = lrelu(BN1(x1)); s = Ws @ xn (8 WMMAs); softmax over K (16
// cols == 16 lanes per half); p = sum_k xn*a -> workspace.
__global__ __launch_bounds__(256) void k_attn(const bf16* __restrict__ x1,
    const float* __restrict__ aff1, const float* __restrict__ Ws,
    float* __restrict__ pout)
{
  __shared__ alignas(64) bf16 wss[OUTc * OUTc];     // 8 KB
  __shared__ float s1s[64], t1s[64];
  __shared__ alignas(64) float xn[8][OUTc * Kk];    // per-wave col-major [col*64+row], 32 KB

  const int lane   = threadIdx.x & 31;
  const int wave   = threadIdx.x >> 5;
  const int gwave  = (int)((blockIdx.x * blockDim.x + threadIdx.x) >> 5);
  const int nwaves = (int)((gridDim.x * blockDim.x) >> 5);
  const int half = lane >> 4, c = lane & 15;

  for (int i = threadIdx.x; i < OUTc * OUTc; i += blockDim.x) wss[i] = (bf16)Ws[i];
  if (threadIdx.x < 64) { s1s[threadIdx.x] = aff1[threadIdx.x]; t1s[threadIdx.x] = aff1[64 + threadIdx.x]; }
  __syncthreads();

  v16bf A[4][2];
#pragma unroll
  for (int t = 0; t < 4; ++t)
#pragma unroll
    for (int kc = 0; kc < 2; ++kc) A[t][kc] = load_a_frag(wss, t * 16, OUTc, kc, lane);

  float* myxn = xn[wave];
  const int colm = lane >> 1;        // my column (0..15), two lanes per column
  const int row0 = (lane & 1) * 32;  // my starting row within the column

  for (int pt = gwave; pt < NPTS; pt += nwaves) {
    const bf16* xt = x1 + (size_t)pt * (OUTc * Kk);
    // x1 is col-major: lane reads 32 contiguous bf16 (4 x b128), applies
    // BN affine + lrelu, writes 32 contiguous f32 to LDS (float4 stores)
    const v8bf* src = (const v8bf*)(xt + lane * 32);
#pragma unroll
    for (int j4 = 0; j4 < 4; ++j4) {
      v8bf raw = src[j4];
      float o[8];
#pragma unroll
      for (int e = 0; e < 8; ++e) {
        int row = row0 + j4 * 8 + e;
        o[e] = lrelu(s1s[row] * (float)raw[e] + t1s[row]);
      }
      float4 lo = make_float4(o[0], o[1], o[2], o[3]);
      float4 hi = make_float4(o[4], o[5], o[6], o[7]);
      *(float4*)(myxn + colm * 64 + row0 + j4 * 8)     = lo;
      *(float4*)(myxn + colm * 64 + row0 + j4 * 8 + 4) = hi;
    }
    wave_lds_fence();

    // B fragments from LDS: float4 loads + cvt to bf16
    v16bf Bf[2];
#pragma unroll
    for (int kc = 0; kc < 2; ++kc) {
      const float4* pc4 = (const float4*)(myxn + c * 64 + kc * 32 + 16 * half);
#pragma unroll
      for (int w = 0; w < 4; ++w) {
        float4 f = pc4[w];
        Bf[kc][w * 4 + 0] = (bf16)f.x; Bf[kc][w * 4 + 1] = (bf16)f.y;
        Bf[kc][w * 4 + 2] = (bf16)f.z; Bf[kc][w * 4 + 3] = (bf16)f.w;
      }
    }

    v8f s[4];
#pragma unroll
    for (int t = 0; t < 4; ++t) {
      v8f a = {};
      a = wmma_bf16(A[t][0], Bf[0], a);
      a = wmma_bf16(A[t][1], Bf[1], a);
      s[t] = a;
    }

    // per-row softmax over 16 cols + attentive pooling (result left in s[t][r],
    // identical across the 16 lanes of each half after the butterflies)
#pragma unroll
    for (int t = 0; t < 4; ++t) {
      const float4* xr = (const float4*)(myxn + c * 64 + t * 16 + 8 * half);
      float4 x0 = xr[0], x1v = xr[1];
      float xv[8] = {x0.x, x0.y, x0.z, x0.w, x1v.x, x1v.y, x1v.z, x1v.w};
#pragma unroll
      for (int r = 0; r < 8; ++r) {
        float v = s[t][r];
        float mx = v;
        for (int m = 1; m < 16; m <<= 1) mx = fmaxf(mx, __shfl_xor(mx, m, 16));
        float e = __expf(v - mx);
        float se = e;
        for (int m = 1; m < 16; m <<= 1) se += __shfl_xor(se, m, 16);
        float pp = (e / se) * xv[r];
        for (int m = 1; m < 16; m <<= 1) pp += __shfl_xor(pp, m, 16);
        s[t][r] = pp;
      }
    }
    // coalesced p store: lane c -> (t = c>>3 and c>>3 + 2, r = c&7)
    {
      const int tt = c >> 3, rr = c & 7;
      pout[(size_t)pt * 64 + tt * 16 + 8 * half + rr]       = s[tt][rr];
      pout[(size_t)pt * 64 + (tt + 2) * 16 + 8 * half + rr] = s[tt + 2][rr];
    }
  }
}

// --------------------------------------------------------------------------
// Kernel D: q = Wm @ P for 16 points per wave (8 WMMAs), accumulate BN2 stats.
__global__ __launch_bounds__(256) void k_gemm3(const float* __restrict__ pin,
    const float* __restrict__ Wm, float* __restrict__ qout,
    float* __restrict__ stats2)
{
  __shared__ alignas(64) bf16 wms[OUTc * OUTc];

  const int lane   = threadIdx.x & 31;
  const int gwave  = (int)((blockIdx.x * blockDim.x + threadIdx.x) >> 5);
  const int nwaves = (int)((gridDim.x * blockDim.x) >> 5);
  const int half = lane >> 4, c = lane & 15;

  for (int i = threadIdx.x; i < OUTc * OUTc; i += blockDim.x) wms[i] = (bf16)Wm[i];
  __syncthreads();

  v16bf A[4][2];
#pragma unroll
  for (int t = 0; t < 4; ++t)
#pragma unroll
    for (int kc = 0; kc < 2; ++kc) A[t][kc] = load_a_frag(wms, t * 16, OUTc, kc, lane);

  float ssum[4][8] = {}, ssq[4][8] = {};
  const int NG = NPTS / 16;

  for (int g = gwave; g < NG; g += nwaves) {
    const float* pb = pin + (size_t)g * 16 * 64;   // 16 points x 64 ch, [pt][ch]
    v16bf Bf[2];
#pragma unroll
    for (int kc = 0; kc < 2; ++kc) {
      const float4* pc4 = (const float4*)(pb + c * 64 + kc * 32 + 16 * half);
#pragma unroll
      for (int w = 0; w < 4; ++w) {
        float4 f = pc4[w];
        Bf[kc][w * 4 + 0] = (bf16)f.x; Bf[kc][w * 4 + 1] = (bf16)f.y;
        Bf[kc][w * 4 + 2] = (bf16)f.z; Bf[kc][w * 4 + 3] = (bf16)f.w;
      }
    }
    v8f q[4];
#pragma unroll
    for (int t = 0; t < 4; ++t) {
      v8f a = {};
      a = wmma_bf16(A[t][0], Bf[0], a);
      a = wmma_bf16(A[t][1], Bf[1], a);
      q[t] = a;
    }
    float* qo = qout + (size_t)g * 16 * 64;
#pragma unroll
    for (int t = 0; t < 4; ++t) {
#pragma unroll
      for (int r = 0; r < 8; ++r) {
        float v = q[t][r];
        ssum[t][r] += v;
        ssq[t][r] += v * v;
      }
      float4 lo = make_float4(q[t][0], q[t][1], q[t][2], q[t][3]);
      float4 hi = make_float4(q[t][4], q[t][5], q[t][6], q[t][7]);
      *(float4*)(qo + c * 64 + t * 16 + 8 * half)     = lo;
      *(float4*)(qo + c * 64 + t * 16 + 8 * half + 4) = hi;
    }
  }

#pragma unroll
  for (int t = 0; t < 4; ++t)
#pragma unroll
    for (int r = 0; r < 8; ++r) {
      float s = ssum[t][r], q2 = ssq[t][r];
      for (int m = 8; m >= 1; m >>= 1) { s += __shfl_xor(s, m, 16); q2 += __shfl_xor(q2, m, 16); }
      if (c == 0) {
        int row = t * 16 + half * 8 + r;
        atomicAdd(&stats2[row], s);
        atomicAdd(&stats2[64 + row], q2);
      }
    }
}

// --------------------------------------------------------------------------
// Kernel F: y = lrelu(BN2(q)), transposed [pt][ch] -> out[(b*64+ch)*N + n]
__global__ __launch_bounds__(256) void k_out(const float* __restrict__ q,
    const float* __restrict__ aff2, float* __restrict__ out)
{
  __shared__ float tile[64 * 64];
  const int blk = blockIdx.x;                 // Bb * (Nn/64) blocks
  const int b  = blk / (Nn / 64);
  const int n0 = (blk % (Nn / 64)) * 64;
  const float* qb = q + ((size_t)b * Nn + n0) * 64;

  for (int i = threadIdx.x; i < 4096; i += 256) {
    int ch = i & 63;
    float v = aff2[ch] * qb[i] + aff2[64 + ch];
    tile[i] = lrelu(v);                       // tile[pidx*64 + ch]
  }
  __syncthreads();
  for (int i = threadIdx.x; i < 4096; i += 256) {
    int ch = i >> 6, pp = i & 63;
    out[((size_t)b * 64 + ch) * Nn + n0 + pp] = tile[pp * 64 + ch];
  }
}

// --------------------------------------------------------------------------
extern "C" void kernel_launch(void* const* d_in, const int* in_sizes, int n_in,
                              void* d_out, int out_size, void* d_ws, size_t ws_size,
                              hipStream_t stream)
{
  (void)in_sizes; (void)n_in; (void)out_size; (void)ws_size;
  const float* xyz  = (const float*)d_in[0];
  const float* feat = (const float*)d_in[1];
  const int*   nidx = (const int*)  d_in[2];
  const float* W1   = (const float*)d_in[3];
  const float* g1   = (const float*)d_in[4];
  const float* b1   = (const float*)d_in[5];
  const float* Ws   = (const float*)d_in[6];
  const float* Wm   = (const float*)d_in[7];
  const float* g2   = (const float*)d_in[8];
  const float* b2   = (const float*)d_in[9];
  float* out = (float*)d_out;

  // workspace layout (~210 MB): stats/affine (2 KB) | p (21 MB) | q (21 MB) | x1 bf16 (168 MB)
  float* stats1 = (float*)d_ws;
  float* aff1   = stats1 + 128;
  float* stats2 = aff1 + 128;
  float* aff2   = stats2 + 128;
  float* pbuf   = aff2 + 128;
  float* qbuf   = pbuf + (size_t)NPTS * 64;
  bf16*  x1     = (bf16*)(qbuf + (size_t)NPTS * 64);

  k_zero <<<1, 512, 0, stream>>>((float*)d_ws);
  k_gemm1<<<1024, 256, 0, stream>>>(xyz, feat, nidx, W1, x1, stats1);
  k_fin  <<<1, 64, 0, stream>>>(stats1, g1, b1, aff1, (float)((size_t)NPTS * Kk));
  k_attn <<<1024, 256, 0, stream>>>(x1, aff1, Ws, pbuf);
  k_gemm3<<<640, 256, 0, stream>>>(pbuf, Wm, qbuf, stats2);
  k_fin  <<<1, 64, 0, stream>>>(stats2, g2, b2, aff2, (float)NPTS);
  k_out  <<<1280, 256, 0, stream>>>(qbuf, aff2, out);
}
